// StatefulCausalHopfieldAttentionHead_12558484374036
// MI455X (gfx1250) — compile-verified, matrix operand order checked by reading, not run on hardware
//
#include <hip/hip_runtime.h>
#include <stdint.h>

// Problem constants (from reference)
#define B_   4
#define L_   2048
#define S_   128
#define DIN_ 1024
#define DH_  128
#define T_   2304          // L + 2S
#define NKT_ 144           // T/16
#define CHUNK_ 72          // T/32
#define SCALE_ 0.08838834764831845f  // 1/sqrt(128)

typedef __attribute__((ext_vector_type(16))) __bf16 v16bf;
typedef __attribute__((ext_vector_type(8)))  float  v8f;

union Frag {
  uint32_t u[8];
  uint4    q[2];
  v16bf    v;
};

__device__ __forceinline__ uint16_t f2bf(float f) {
  union { float f; uint32_t u; } a; a.f = f;
  uint32_t u = a.u + 0x7FFFu + ((a.u >> 16) & 1u);   // round-to-nearest-even
  return (uint16_t)(u >> 16);
}
__device__ __forceinline__ uint32_t pack2bf(float lo, float hi) {
  return (uint32_t)f2bf(lo) | ((uint32_t)f2bf(hi) << 16);
}
__device__ __forceinline__ v8f bfwmma(const Frag& a, const Frag& b, v8f c) {
  return __builtin_amdgcn_wmma_f32_16x16x32_bf16(false, a.v, false, b.v,
                                                 (short)0, c, false, false);
}

// ---------------- conversion / packing kernels ----------------

__global__ void cvt_bf16_kernel(const float* __restrict__ in,
                                uint16_t* __restrict__ out, int n) {
  int i = blockIdx.x * blockDim.x + threadIdx.x;
  if (i < n) out[i] = f2bf(in[i]);
}

// cope_emb is (DH, T) row-major; store transposed (T, DH) bf16 so the
// B-fragment of q @ cope is two contiguous 16B loads per lane.
__global__ void cope_transpose_kernel(const float* __restrict__ cope,
                                      uint16_t* __restrict__ copeT) {
  int i = blockIdx.x * blockDim.x + threadIdx.x;
  if (i >= T_ * DH_) return;
  int t = i / DH_, d = i - t * DH_;
  copeT[i] = f2bf(cope[d * T_ + t]);
}

// Pack a (kd x 128) fp32 weight into the CDNA5 16-bit B-matrix VGPR layout:
// index ((ntile*nks + ks)*32 + lane)*8 + e  (uint32 = packed bf16 pair)
//   lanes 0-15: K = kb..kb+7 (e=0..3) and 16+kb.. (e=4..7), kb = (lane>>4)*8
__global__ void pack_w_kernel(const float* __restrict__ W,
                              uint32_t* __restrict__ P, int nks) {
  int i = blockIdx.x * blockDim.x + threadIdx.x;
  int total = 8 * nks * 32 * 8;
  if (i >= total) return;
  int idx = i;
  int e    = idx & 7;  idx >>= 3;
  int lane = idx & 31; idx >>= 5;
  int ks   = idx % nks;
  int ntl  = idx / nks;
  int n  = ntl * 16 + (lane & 15);
  int kb = (lane >> 4) * 8;
  int k  = ks * 32 + ((e < 4) ? (kb + 2 * e) : (16 + kb + 2 * (e - 4)));
  P[i] = pack2bf(W[k * 128 + n], W[(k + 1) * 128 + n]);
}

// ---------------- projection GEMM (M x kd) @ (kd x 128) -------------
// One block = one 16-row tile of M; 8 waves = 8 column tiles (N=128).
// Per-row-tile weight select: prefix/suffix rows (t<S or t>=S+L) use Pspec.
__global__ __launch_bounds__(256)
void proj_gemm_kernel(const uint16_t* __restrict__ A, int kd,
                      const uint32_t* __restrict__ Pmain,
                      const uint32_t* __restrict__ Pspec,
                      uint16_t* __restrict__ outRM,   // bf16 row-major (M,128) or null
                      uint16_t* __restrict__ outTV) { // bf16 transposed (B,128,T) or null
  int m0   = blockIdx.x * 16;
  int lane = threadIdx.x & 31;
  int wid  = threadIdx.x >> 5;      // ntile 0..7
  int nks  = kd >> 5;
  int t    = m0 % T_;
  const uint32_t* P = ((t < S_) || (t >= S_ + L_)) ? Pspec : Pmain;

  int arow = lane & 15;
  int kb   = (lane >> 4) * 8;
  const uint16_t* ap = A + (size_t)(m0 + arow) * kd + kb;
  const uint32_t* bp = P + (size_t)wid * nks * 256 + lane * 8;

  v8f c = {};
  for (int ks = 0; ks < nks; ++ks) {
    Frag fa, fb;
    fa.q[0] = *(const uint4*)(ap + ks * 32);
    fa.q[1] = *(const uint4*)(ap + ks * 32 + 16);
    fb.q[0] = *(const uint4*)(bp + ks * 256);
    fb.q[1] = *(const uint4*)(bp + ks * 256 + 4);
    c = bfwmma(fa, fb, c);
  }

  int col  = wid * 16 + (lane & 15);
  int rsel = (lane >> 4) * 8;
  if (outRM) {
#pragma unroll
    for (int r = 0; r < 8; ++r)
      outRM[(size_t)(m0 + r + rsel) * DH_ + col] = f2bf(c[r]);
  }
  if (outTV) {
#pragma unroll
    for (int r = 0; r < 8; ++r) {
      int row = m0 + r + rsel;
      int b   = row / T_;
      int tt  = row - b * T_;
      outTV[((size_t)b * DH_ + col) * T_ + tt] = f2bf(c[r]);
    }
  }
}

// ---------------- fused attention (one 16-query tile per WG) -------------
// LDS: logits[16][T] + logits_int[16][T] fp32 (288 KB dynamic, fits 320KB WGP LDS)
__global__ __launch_bounds__(256)
void attn_kernel(const uint16_t* __restrict__ Qb,   // (B,T,DH) bf16
                 const uint16_t* __restrict__ Kb,   // (B,T,DH) bf16
                 const uint16_t* __restrict__ Vt,   // (B,DH,T) bf16
                 const uint16_t* __restrict__ copeT,// (T,DH) bf16
                 uint16_t* __restrict__ fullBF,     // (B,T,DH) bf16 or null
                 float*    __restrict__ outF32) {   // (B,T,DH) fp32 or null
  extern __shared__ float smem[];
  float* lg = smem;            // 16 * T
  float* li = smem + 16 * T_;  // 16 * T
  __shared__ float rowscale[16];

  int qt   = blockIdx.x;       // 0..143
  int b    = blockIdx.y;       // 0..3
  int lane = threadIdx.x & 31;
  int wid  = threadIdx.x >> 5;
  int qr0  = qt * 16;

  const uint16_t* Qbase = Qb + (size_t)b * T_ * DH_;
  const uint16_t* Kbase = Kb + (size_t)b * T_ * DH_;
  const uint16_t* Vbase = Vt + (size_t)b * DH_ * T_;

  int arow = lane & 15;
  int kb   = (lane >> 4) * 8;
  int rsel = (lane >> 4) * 8;
  int col  = lane & 15;

  // Q row-tile fragments (reused for all 288 column tiles)
  Frag fa[4];
  {
    const uint16_t* qp = Qbase + (size_t)(qr0 + arow) * DH_ + kb;
#pragma unroll
    for (int ks = 0; ks < 4; ++ks) {
      fa[ks].q[0] = *(const uint4*)(qp + ks * 32);
      fa[ks].q[1] = *(const uint4*)(qp + ks * 32 + 16);
    }
  }

  // Phase 1: logits = q @ k^T  (row-major K makes B-frag loads contiguous)
  for (int j = 0; j < NKT_ / 8; ++j) {
    int kt = wid + 8 * j;
    const uint16_t* kp = Kbase + (size_t)(kt * 16 + arow) * DH_ + kb;
    v8f c = {};
#pragma unroll
    for (int ks = 0; ks < 4; ++ks) {
      Frag fb;
      fb.q[0] = *(const uint4*)(kp + ks * 32);
      fb.q[1] = *(const uint4*)(kp + ks * 32 + 16);
      c = bfwmma(fa[ks], fb, c);
    }
#pragma unroll
    for (int r = 0; r < 8; ++r)
      lg[(r + rsel) * T_ + kt * 16 + col] = c[r];
  }

  // Phase 2: logits_int = q @ cope_emb (transposed cope -> contiguous loads)
  for (int j = 0; j < NKT_ / 8; ++j) {
    int tt = wid + 8 * j;
    const uint16_t* cp = copeT + (size_t)(tt * 16 + arow) * DH_ + kb;
    v8f c = {};
#pragma unroll
    for (int ks = 0; ks < 4; ++ks) {
      Frag fb;
      fb.q[0] = *(const uint4*)(cp + ks * 32);
      fb.q[1] = *(const uint4*)(cp + ks * 32 + 16);
      c = bfwmma(fa[ks], fb, c);
    }
#pragma unroll
    for (int r = 0; r < 8; ++r)
      li[(r + rsel) * T_ + tt * 16 + col] = c[r];
  }
  __syncthreads();

  // Phase 3: per-row suffix-scan of sigmoid gates (CoPE pos), bias gather,
  // causal mask + scaled logits, two-pass softmax (unnormalized exp kept in
  // LDS; 1/rowsum folded into the final output). Wave w owns rows 2w, 2w+1;
  // lane owns a contiguous 72-element chunk; wave32 shuffle scans/reductions.
  for (int rr = 0; rr < 2; ++rr) {
    int r  = wid * 2 + rr;
    int qg = qr0 + r;
    float* row = lg + r * T_;
    const float* lir = li + r * T_;
    int e0 = lane * CHUNK_;

    float csum = 0.f;
    for (int e = 0; e < CHUNK_; ++e) {
      float x = row[e0 + e];
      csum += 1.0f / (1.0f + __expf(-x));
    }
    // inclusive suffix sum of chunk sums across 32 lanes
    float inc = csum;
#pragma unroll
    for (int off = 1; off < 32; off <<= 1) {
      float tmo = __shfl_down(inc, off, 32);
      if (lane + off < 32) inc += tmo;
    }
    float run = inc - csum;   // exclusive suffix (lanes > me)

    float mx = -3.0e38f;
    for (int e = CHUNK_ - 1; e >= 0; --e) {
      int kidx = e0 + e;
      float x = row[kidx];
      float g = 1.0f / (1.0f + __expf(-x));
      run += g;                                   // pos = suffix-cumsum
      float pos = fminf(run, (float)(T_ - 1));
      float pf  = floorf(pos);
      float w   = pos - pf;
      int ipf = (int)pf;
      int ipc = (int)ceilf(pos);
      float bias = lir[ipc] * w + lir[ipf] * (1.0f - w);
      float val  = (kidx <= qg) ? (SCALE_ * x + bias) : -3.0e38f;
      row[kidx] = val;
      mx = fmaxf(mx, val);
    }
#pragma unroll
    for (int off = 16; off > 0; off >>= 1)
      mx = fmaxf(mx, __shfl_xor(mx, off, 32));

    float sm = 0.f;
    for (int e = 0; e < CHUNK_; ++e) {
      int kidx = e0 + e;
      float ev = __expf(row[kidx] - mx);
      row[kidx] = ev;                             // unnormalized attn
      sm += ev;
    }
#pragma unroll
    for (int off = 16; off > 0; off >>= 1)
      sm += __shfl_xor(sm, off, 32);
    if (lane == 0) rowscale[r] = 1.0f / sm;
  }
  __syncthreads();

  // Phase 4: out = attn @ V. A-frag packed from LDS fp32 (ds_load_b128 x4),
  // B-frag from transposed V (contiguous). Wave w = output column tile w.
  v8f c = {};
  for (int ks = 0; ks < T_ / 32; ++ks) {
    Frag fav, fbv;
    const float* ap = lg + arow * T_ + ks * 32 + kb;
    float4 a0 = *(const float4*)(ap);
    float4 a1 = *(const float4*)(ap + 4);
    float4 a2 = *(const float4*)(ap + 16);
    float4 a3 = *(const float4*)(ap + 20);
    fav.u[0] = pack2bf(a0.x, a0.y); fav.u[1] = pack2bf(a0.z, a0.w);
    fav.u[2] = pack2bf(a1.x, a1.y); fav.u[3] = pack2bf(a1.z, a1.w);
    fav.u[4] = pack2bf(a2.x, a2.y); fav.u[5] = pack2bf(a2.z, a2.w);
    fav.u[6] = pack2bf(a3.x, a3.y); fav.u[7] = pack2bf(a3.z, a3.w);
    const uint16_t* vp = Vbase + (size_t)(wid * 16 + arow) * T_ + ks * 32 + kb;
    fbv.q[0] = *(const uint4*)(vp);
    fbv.q[1] = *(const uint4*)(vp + 16);
    c = bfwmma(fav, fbv, c);
  }
#pragma unroll
  for (int r = 0; r < 8; ++r) {
    float o = c[r] * rowscale[r + rsel];
    int row = qr0 + r + rsel;
    size_t oi = ((size_t)b * T_ + row) * DH_ + wid * 16 + col;
    if (outF32) outF32[oi] = o;
    if (fullBF) fullBF[oi] = f2bf(o);
  }
}

// ---------------- host orchestration ----------------

extern "C" void kernel_launch(void* const* d_in, const int* in_sizes, int n_in,
                              void* d_out, int out_size, void* d_ws, size_t ws_size,
                              hipStream_t stream) {
  const float* x    = (const float*)d_in[0];
  const float* Wh   = (const float*)d_in[1];
  const float* Whs  = (const float*)d_in[2];
  const float* Wq   = (const float*)d_in[3];
  const float* Wk   = (const float*)d_in[4];
  const float* Wv   = (const float*)d_in[5];
  const float* Wqs  = (const float*)d_in[6];
  const float* Wks  = (const float*)d_in[7];
  const float* Wvs  = (const float*)d_in[8];
  const float* cope = (const float*)d_in[9];

  char* ws = (char*)d_ws;
  size_t off = 0;
  auto alloc = [&](size_t bytes) -> char* {
    char* p = ws + off;
    off += (bytes + 255) & ~(size_t)255;
    return p;
  };
  uint16_t* xb    = (uint16_t*)alloc((size_t)B_ * T_ * DIN_ * 2);
  uint32_t* Whp   = (uint32_t*)alloc((size_t)8 * 32 * 256 * 4);
  uint32_t* Whsp  = (uint32_t*)alloc((size_t)8 * 32 * 256 * 4);
  uint32_t* Wqp   = (uint32_t*)alloc((size_t)8 * 4 * 256 * 4);
  uint32_t* Wkp   = (uint32_t*)alloc((size_t)8 * 4 * 256 * 4);
  uint32_t* Wvp   = (uint32_t*)alloc((size_t)8 * 4 * 256 * 4);
  uint32_t* Wqsp  = (uint32_t*)alloc((size_t)8 * 4 * 256 * 4);
  uint32_t* Wksp  = (uint32_t*)alloc((size_t)8 * 4 * 256 * 4);
  uint32_t* Wvsp  = (uint32_t*)alloc((size_t)8 * 4 * 256 * 4);
  uint16_t* copeT = (uint16_t*)alloc((size_t)T_ * DH_ * 2);
  uint16_t* Hb    = (uint16_t*)alloc((size_t)B_ * T_ * DH_ * 2);
  uint16_t* Kbuf  = (uint16_t*)alloc((size_t)B_ * T_ * DH_ * 2);
  uint16_t* Vtb   = (uint16_t*)alloc((size_t)B_ * DH_ * T_ * 2);
  uint16_t* Qbuf  = (uint16_t*)alloc((size_t)B_ * T_ * DH_ * 2);
  uint16_t* Fullb = (uint16_t*)alloc((size_t)B_ * T_ * DH_ * 2);
  (void)ws_size; (void)in_sizes; (void)n_in; (void)out_size;

  // conversions + packing
  {
    int n = B_ * T_ * DIN_;
    cvt_bf16_kernel<<<(n + 255) / 256, 256, 0, stream>>>(x, xb, n);
  }
  pack_w_kernel<<<(8 * 32 * 256) / 256, 256, 0, stream>>>(Wh,  Whp,  32);
  pack_w_kernel<<<(8 * 32 * 256) / 256, 256, 0, stream>>>(Whs, Whsp, 32);
  pack_w_kernel<<<(8 * 4 * 256) / 256, 256, 0, stream>>>(Wq,  Wqp,  4);
  pack_w_kernel<<<(8 * 4 * 256) / 256, 256, 0, stream>>>(Wk,  Wkp,  4);
  pack_w_kernel<<<(8 * 4 * 256) / 256, 256, 0, stream>>>(Wv,  Wvp,  4);
  pack_w_kernel<<<(8 * 4 * 256) / 256, 256, 0, stream>>>(Wqs, Wqsp, 4);
  pack_w_kernel<<<(8 * 4 * 256) / 256, 256, 0, stream>>>(Wks, Wksp, 4);
  pack_w_kernel<<<(8 * 4 * 256) / 256, 256, 0, stream>>>(Wvs, Wvsp, 4);
  cope_transpose_kernel<<<(T_ * DH_ + 255) / 256, 256, 0, stream>>>(cope, copeT);

  const int Mtiles = B_ * T_ / 16;  // 576
  // hidden + fixed K, V projections
  proj_gemm_kernel<<<Mtiles, 256, 0, stream>>>(xb, DIN_, Whp, Whsp, Hb, nullptr);
  proj_gemm_kernel<<<Mtiles, 256, 0, stream>>>(Hb, DH_, Wkp, Wksp, Kbuf, nullptr);
  proj_gemm_kernel<<<Mtiles, 256, 0, stream>>>(Hb, DH_, Wvp, Wvsp, nullptr, Vtb);
  // iteration 1
  proj_gemm_kernel<<<Mtiles, 256, 0, stream>>>(Hb, DH_, Wqp, Wqsp, Qbuf, nullptr);
  dim3 ag(NKT_, B_);
  size_t lds = (size_t)16 * T_ * 4 * 2;  // 288 KB dynamic LDS
  attn_kernel<<<ag, 256, lds, stream>>>(Qbuf, Kbuf, Vtb, copeT, Fullb, nullptr);
  // iteration 2 (Q from iter-1 output; K/V unchanged)
  proj_gemm_kernel<<<Mtiles, 256, 0, stream>>>(Fullb, DH_, Wqp, Wqsp, Qbuf, nullptr);
  attn_kernel<<<ag, 256, lds, stream>>>(Qbuf, Kbuf, Vtb, copeT, nullptr, (float*)d_out);
}
